// EchoDynamicsNet_2413771621070
// MI455X (gfx1250) — matrix operand order repaired and checked by reading, not compiled, not run on hardware
//
#include <hip/hip_runtime.h>

// ---------------------------------------------------------------------------
// EchoDynamicsNet forward for MI455X (gfx1250, wave32, WMMA).
// B=16, L=512, D=768, H=768, NH=12, HD=64, FD=1536, CD=256, K=2, KS=5, NC=8
// Heavy GEMMs -> v_wmma_f32_16x16x32_f16 (fp32 accum).  LSTM scan keeps
// h/c/gates in 264KB WGP LDS; attention stages scores/att in 192KB LDS.
// ---------------------------------------------------------------------------

typedef _Float16 h8  __attribute__((ext_vector_type(8)));
typedef _Float16 h16 __attribute__((ext_vector_type(16)));
typedef float    f8  __attribute__((ext_vector_type(8)));

static __device__ __forceinline__ f8 f8zero() {
  f8 z = {0.f, 0.f, 0.f, 0.f, 0.f, 0.f, 0.f, 0.f};
  return z;
}

static __device__ __forceinline__ f8 wmma32(h16 a, h16 b, f8 c) {
  return __builtin_amdgcn_wmma_f32_16x16x32_f16(false, a, false, b, (short)0, c,
                                                false, false);
}

// A-fragment (16x32 f16, contiguous k = 16*half + s). p already includes (lane>>4)*16.
static __device__ __forceinline__ h16 ld_a16(const _Float16* p) {
  h8 lo = *(const h8*)p;
  h8 hi = *(const h8*)(p + 8);
  h16 r;
#pragma unroll
  for (int i = 0; i < 8; ++i) { r[i] = lo[i]; r[i + 8] = hi[i]; }
  return r;
}

static __device__ __forceinline__ long lmin(long a, long b) { return a < b ? a : b; }

static __device__ __forceinline__ long remap(long r, long dv, long sk, long of) {
  return (r / dv) * sk + (r % dv) + of;
}

static __device__ __forceinline__ float sigf(float x) {
  return 1.f / (1.f + __expf(-x));
}

#define BIGDIV ((long)1 << 40)

// ---------------------------------------------------------------------------
// Generic WMMA GEMM:  C[M,N] = act(A[M,K] @ W[N,K]^T + bias [+ resid])
// Row remap (r -> (r/div)*skip + r%div + off) on A and C for strided slices.
// 256 threads = 8 waves; block tile 64x128; wave tile 32x32.
// ---------------------------------------------------------------------------
__global__ void __launch_bounds__(256) gemm_wmma_kernel(
    const _Float16* __restrict__ A, int lda, long Adiv, long Askip, long Aoff,
    const _Float16* __restrict__ W, int ldw, const float* __restrict__ bias,
    const float* __restrict__ resid, int ldr, float* outF, _Float16* outH,
    int ldc, long Cdiv, long Cskip, long Coff, int M, int N, int K, int act) {
  const int tid = threadIdx.x, lane = tid & 31, wid = tid >> 5;
  const int wm = wid & 1, wn = wid >> 1;
  const int mlo = lane & 15, half = lane >> 4;
  const long m00 = (long)blockIdx.y * 64 + wm * 32;
  const long n00 = (long)blockIdx.x * 128 + wn * 32;

  const long rA0 = lmin(m00 + mlo, (long)M - 1);
  const long rA1 = lmin(m00 + 16 + mlo, (long)M - 1);
  const _Float16* pa0 = A + remap(rA0, Adiv, Askip, Aoff) * (long)lda + half * 16;
  const _Float16* pa1 = A + remap(rA1, Adiv, Askip, Aoff) * (long)lda + half * 16;
  const long cB0 = lmin(n00 + mlo, (long)N - 1);
  const long cB1 = lmin(n00 + 16 + mlo, (long)N - 1);
  const _Float16* pb0 = W + cB0 * (long)ldw + half * 16;
  const _Float16* pb1 = W + cB1 * (long)ldw + half * 16;

  f8 c00 = f8zero(), c01 = f8zero(), c10 = f8zero(), c11 = f8zero();
  for (int k0 = 0; k0 < K; k0 += 32) {
    h16 a0 = ld_a16(pa0 + k0);
    h16 a1 = ld_a16(pa1 + k0);
    h16 b0 = *(const h16*)(pb0 + k0);
    h16 b1 = *(const h16*)(pb1 + k0);
    c00 = wmma32(a0, b0, c00);
    c01 = wmma32(a0, b1, c01);
    c10 = wmma32(a1, b0, c10);
    c11 = wmma32(a1, b1, c11);
  }

#pragma unroll
  for (int t = 0; t < 4; ++t) {
    f8 cc = (t == 0) ? c00 : (t == 1) ? c01 : (t == 2) ? c10 : c11;
    const long mb = m00 + (t >> 1) * 16;
    const long nb = n00 + (t & 1) * 16;
#pragma unroll
    for (int r = 0; r < 8; ++r) {
      long row = mb + r + half * 8;
      long col = nb + mlo;
      if (row < M && col < N) {
        long cr = remap(row, Cdiv, Cskip, Coff);
        float v = cc[r] + bias[col];
        if (resid) v += resid[cr * (long)ldr + col];
        if (act == 1) v = fmaxf(v, 0.f);
        else if (act == 2) v = v * sigf(1.702f * v);
        if (outF) outF[cr * (long)ldc + col] = v;
        if (outH) outH[cr * (long)ldc + col] = (_Float16)v;
      }
    }
  }
}

// ---------------------------------------------------------------------------
// Bidirectional LSTM scan: one 1024-thread WG per direction.
// LDS: h f16 16x768 | c f32 16x768 | gates f32 16x3072  (264 KB)
// ---------------------------------------------------------------------------
__global__ void __launch_bounds__(1024) lstm_scan_kernel(
    const _Float16* __restrict__ xpF, const _Float16* __restrict__ xpR,
    const _Float16* __restrict__ whhF, const _Float16* __restrict__ whhR,
    const float* __restrict__ h0, const float* __restrict__ c0,
    _Float16* __restrict__ out, _Float16* __restrict__ hT) {
  extern __shared__ char smem[];
  _Float16* hs = (_Float16*)smem;
  float* cs = (float*)(smem + 16 * 768 * 2);
  float* gt = (float*)(smem + 16 * 768 * 2 + 16 * 768 * 4);

  const int dir = blockIdx.x;
  const _Float16* xp = dir ? xpR : xpF;
  const _Float16* whh = dir ? whhR : whhF;
  const int tid = threadIdx.x, lane = tid & 31, wid = tid >> 5;
  const int mlo = lane & 15, half = lane >> 4;

  for (int i = tid; i < 16 * 768; i += 1024) {
    hs[i] = (_Float16)h0[i];
    cs[i] = c0[i];
  }
  __syncthreads();

  for (int t = 0; t < 511; ++t) {
    const int tt = dir ? (510 - t) : t;
    for (int it = 0; it < 6; ++it) {          // 192 n-tiles over 32 waves
      const int n0 = (wid + it * 32) * 16;
      f8 acc = f8zero();
      const _Float16* hb = hs + mlo * 768 + half * 16;
      const _Float16* wb = whh + (long)(n0 + mlo) * 768 + half * 16;
      for (int k0 = 0; k0 < 768; k0 += 32) {
        h16 a = ld_a16(hb + k0);
        h16 b = *(const h16*)(wb + k0);
        acc = wmma32(a, b, acc);
      }
#pragma unroll
      for (int r = 0; r < 8; ++r) {
        const int bb = r + half * 8;
        const int col = n0 + mlo;
        float xv = (float)xp[((long)bb * 512 + tt + 1) * 3072 + col];
        gt[bb * 3072 + col] = acc[r] + xv;
      }
    }
    __syncthreads();
    for (int i = tid; i < 16 * 768; i += 1024) {
      const int bb = i / 768, j = i - bb * 768;
      const float* gr = gt + (long)bb * 3072;
      float gi = gr[j], gf = gr[768 + j], gg = gr[1536 + j], go = gr[2304 + j];
      float cc = sigf(gf) * cs[i] + sigf(gi) * tanhf(gg);
      float hh = sigf(go) * tanhf(cc);
      cs[i] = cc;
      hs[i] = (_Float16)hh;
      out[((long)bb * 511 + tt) * 1536 + dir * 768 + j] = (_Float16)hh;
      if (t == 510) hT[(long)bb * 1536 + dir * 768 + j] = (_Float16)hh;
    }
    __syncthreads();
  }
}

// ---------------------------------------------------------------------------
// Attention: block per (b, head, 64-row q-tile). 128 threads = 4 waves.
// LDS: scores f32 64x512 + att f16 64x512 = 192 KB
// ---------------------------------------------------------------------------
__global__ void __launch_bounds__(128) attn_kernel(
    const _Float16* __restrict__ qkv, const _Float16* __restrict__ vT,
    const int* __restrict__ mask, _Float16* __restrict__ o) {
  extern __shared__ char smem[];
  float* sc = (float*)smem;
  _Float16* at = (_Float16*)(smem + 64 * 512 * 4);

  const int tid = threadIdx.x, lane = tid & 31, wid = tid >> 5;
  const int bid = blockIdx.x;
  const int qt = bid & 7;
  const int h = (bid >> 3) % 12;
  const int b = bid / 96;
  const int mlo = lane & 15, half = lane >> 4;

  const long qrow = (long)qt * 64 + wid * 16 + mlo;
  const _Float16* qbase = qkv + ((long)b * 512 + qrow) * 2304 + h * 64 + half * 16;
  for (int j0 = 0; j0 < 512; j0 += 16) {
    const _Float16* kbase =
        qkv + ((long)b * 512 + j0 + mlo) * 2304 + 768 + h * 64 + half * 16;
    f8 acc = f8zero();
#pragma unroll
    for (int k0 = 0; k0 < 64; k0 += 32) {
      h16 a = ld_a16(qbase + k0);
      h16 bb = *(const h16*)(kbase + k0);
      acc = wmma32(a, bb, acc);
    }
#pragma unroll
    for (int r = 0; r < 8; ++r) {
      const int qr = wid * 16 + r + half * 8;
      const int j = j0 + mlo;
      float s = acc[r] * 0.125f;
      if (mask[b * 512 + j] == 0) s = -1e9f;
      sc[qr * 512 + j] = s;
    }
  }
  __syncthreads();
  if (tid < 64) {
    float* row = sc + tid * 512;
    float mx = -3.4e38f;
    for (int j = 0; j < 512; ++j) mx = fmaxf(mx, row[j]);
    float sum = 0.f;
    for (int j = 0; j < 512; ++j) {
      float e = __expf(row[j] - mx);
      row[j] = e;
      sum += e;
    }
    float inv = 1.f / sum;
    _Float16* ar = at + tid * 512;
    for (int j = 0; j < 512; ++j) ar[j] = (_Float16)(row[j] * inv);
  }
  __syncthreads();
  const _Float16* abase = at + (wid * 16 + mlo) * 512 + half * 16;
  for (int ni = 0; ni < 4; ++ni) {
    const int d0 = ni * 16;
    const _Float16* vb = vT + (((long)b * 12 + h) * 64 + d0 + mlo) * 512 + half * 16;
    f8 acc = f8zero();
    for (int k0 = 0; k0 < 512; k0 += 32) {
      h16 a = ld_a16(abase + k0);
      h16 bb = *(const h16*)(vb + k0);
      acc = wmma32(a, bb, acc);
    }
#pragma unroll
    for (int r = 0; r < 8; ++r) {
      const long q = (long)qt * 64 + wid * 16 + r + half * 8;
      const int col = h * 64 + d0 + mlo;
      o[((long)b * 512 + q) * 768 + col] = (_Float16)acc[r];
    }
  }
}

// ---------------------------------------------------------------------------
// Helper kernels
// ---------------------------------------------------------------------------
__global__ void cvt_f32_f16(const float* __restrict__ x, _Float16* __restrict__ y,
                            long n) {
  long i = (long)blockIdx.x * blockDim.x + threadIdx.x;
  long st = (long)gridDim.x * blockDim.x;
  for (; i < n; i += st) y[i] = (_Float16)x[i];
}

__global__ void vadd_kernel(const float* a, const float* b, float* o, int n) {
  int i = blockIdx.x * blockDim.x + threadIdx.x;
  if (i < n) o[i] = a[i] + b[i];
}

__global__ void layernorm_kernel(const float* __restrict__ x,
                                 const float* __restrict__ g,
                                 const float* __restrict__ bt, int rows, int W,
                                 float* outF, _Float16* outH, int act) {
  const int wid = threadIdx.x >> 5, lane = threadIdx.x & 31;
  const int row = blockIdx.x * 8 + wid;
  if (row >= rows) return;
  const float* xr = x + (long)row * W;
  float s = 0.f, ss = 0.f;
  for (int i = lane; i < W; i += 32) {
    float v = xr[i];
    s += v;
    ss += v * v;
  }
  for (int off = 16; off; off >>= 1) {
    s += __shfl_xor(s, off, 32);
    ss += __shfl_xor(ss, off, 32);
  }
  const float mean = s / W;
  const float var = ss / W - mean * mean;
  const float rs = rsqrtf(var + 1e-5f);
  for (int i = lane; i < W; i += 32) {
    float v = (xr[i] - mean) * rs * g[i] + bt[i];
    if (act == 1) v = fmaxf(v, 0.f);
    if (outF) outF[(long)row * W + i] = v;
    if (outH) outH[(long)row * W + i] = (_Float16)v;
  }
}

__global__ void cls_gate_kernel(const float* X, const float* gate,
                                const float* xhat, float* Xf) {
  int i = blockIdx.x * blockDim.x + threadIdx.x;
  if (i >= 16 * 768) return;
  int b = i / 768, j = i - b * 768;
  float gg = sigf(gate[j]);
  Xf[((long)b * 512) * 768 + j] =
      X[((long)b * 512) * 768 + j] * gg + xhat[i] * (1.f - gg);
}

__global__ void vtrans_kernel(const _Float16* __restrict__ qkv,
                              _Float16* __restrict__ vT) {
  long i = (long)blockIdx.x * blockDim.x + threadIdx.x;
  if (i >= (long)16 * 12 * 64 * 512) return;
  int j = (int)(i & 511);
  long r = i >> 9;
  int d = (int)(r & 63);
  r >>= 6;
  int h = (int)(r % 12);
  int b = (int)(r / 12);
  vT[i] = qkv[((long)b * 512 + j) * 2304 + 1536 + h * 64 + d];
}

__global__ void zprep_kernel(const float* __restrict__ Xeen,
                             const float* __restrict__ alpha,
                             _Float16* __restrict__ Zpre) {
  int i = blockIdx.x * blockDim.x + threadIdx.x;
  if (i >= 16 * 768) return;
  int b = i / 768, j = i - b * 768;
  float m = fmaxf(alpha[0], alpha[1]);
  float e0 = __expf(alpha[0] - m), e1 = __expf(alpha[1] - m);
  float a0 = e0 / (e0 + e1), a1 = e1 / (e0 + e1);
  const float* base = Xeen + ((long)b * 512) * 768 + j;
  float s0 = base[(long)1 * 768], s1 = base[(long)2 * 768];
  float s509 = base[(long)510 * 768], s510 = base[(long)511 * 768];
  float v = (a0 * (s510 - s0) + a1 * (s509 + s510 - s0 - s1)) * (1.f / 510.f);
  Zpre[i] = (_Float16)v;
}

__global__ void softmax2_kernel(const float* logits, float* om) {
  int b = blockIdx.x * blockDim.x + threadIdx.x;
  if (b >= 16) return;
  float a = logits[2 * b], c = logits[2 * b + 1];
  float m = fmaxf(a, c);
  float e0 = __expf(a - m), e1 = __expf(c - m);
  float s = e0 + e1;
  om[2 * b] = e0 / s;
  om[2 * b + 1] = e1 / s;
}

__global__ void yc0_kernel(const float* __restrict__ V3, const int* __restrict__ mask,
                           const float* __restrict__ om,
                           const float* __restrict__ wbank,
                           const float* __restrict__ bbank, float* __restrict__ Yf,
                           _Float16* __restrict__ Yh) {
  int idx = blockIdx.x * blockDim.x + threadIdx.x;
  if (idx >= 16 * 256) return;
  int b = idx >> 8, o = idx & 255;
  float om0 = om[2 * b], om1 = om[2 * b + 1];
  float acc = om0 * bbank[o] + om1 * bbank[256 + o];
  for (int l = 0; l < 3; ++l) {
    float mk = mask[b * 512 + l] ? 1.f : 0.f;
    const float* vrow = V3 + ((long)b * 3 + l) * 256;
    int j = l + 2;
    const float* w0 = wbank + ((long)o * 256) * 5 + j;
    const float* w1 = wbank + ((long)(256 + o) * 256) * 5 + j;
    for (int c = 0; c < 256; ++c) {
      float v = vrow[c] * mk;
      acc += v * (om0 * w0[c * 5] + om1 * w1[c * 5]);
    }
  }
  Yf[idx] = acc;
  Yh[idx] = (_Float16)acc;
}

__global__ void fuse0_kernel(const float* Xeen, const float* Xconv, const float* fg,
                             _Float16* out) {
  int i = blockIdx.x * blockDim.x + threadIdx.x;
  if (i >= 16 * 768) return;
  int b = i / 768, j = i - b * 768;
  float g = sigf(fg[j]);
  float v = Xeen[((long)b * 512) * 768 + j] * g + Xconv[i] * (1.f - g);
  out[i] = (_Float16)v;
}

// ---------------------------------------------------------------------------
// Host
// ---------------------------------------------------------------------------
static void gemm(hipStream_t s, const _Float16* A, int lda, long Adiv, long Askip,
                 long Aoff, const _Float16* W, int ldw, const float* bias,
                 const float* resid, int ldr, float* outF, _Float16* outH, int ldc,
                 long Cdiv, long Cskip, long Coff, int M, int N, int K, int act) {
  dim3 g((N + 127) / 128, (M + 63) / 64);
  gemm_wmma_kernel<<<g, 256, 0, s>>>(A, lda, Adiv, Askip, Aoff, W, ldw, bias, resid,
                                     ldr, outF, outH, ldc, Cdiv, Cskip, Coff, M, N,
                                     K, act);
}

extern "C" void kernel_launch(void* const* d_in, const int* in_sizes, int n_in,
                              void* d_out, int out_size, void* d_ws, size_t ws_size,
                              hipStream_t stream) {
  (void)in_sizes; (void)n_in; (void)out_size; (void)ws_size;
  const float* X = (const float*)d_in[0];
  const int* mask = (const int*)d_in[1];
  const float* P[57];
  for (int i = 0; i < 57; ++i) P[i] = (const float*)d_in[2 + i];
  enum {
    h0_w = 0, h0_b, c0_w, c0_b, wih_f, whh_f, bih_f, bhh_f, wih_r, whh_r, bih_r,
    bhh_r, seqp_w, seqp_b, cls_gate, ln1_g, ln1_b, qkv_w, qkv_b, out_w, out_b,
    ln2_g, ln2_b, mlp1_w, mlp1_b, mlp2_w, mlp2_b, ffn1_w, ffn1_b, ffn2_w, ffn2_b,
    lno_g, lno_b, alpha, zproj_w, zproj_b, cp_w, cp_b, zin_w, zin_b, zw1_w, zw1_b,
    zw2_w, zw2_b, wbank, bbank, cb_w, cb_b, fuse_gate, lnf_g, lnf_b, c1_w, c1_b,
    cln_g, cln_b, c2_w, c2_b
  };

  char* base = (char*)d_ws;
  size_t cur = 0;
  auto alloc = [&](size_t bytes) -> void* {
    size_t a = (cur + 255) & ~(size_t)255;
    cur = a + bytes;
    return (void*)(base + a);
  };
  auto cvtw = [&](const float* src, long n) -> _Float16* {
    _Float16* d = (_Float16*)alloc((size_t)n * 2);
    long blocks = (n + 255) / 256;
    if (blocks > 8192) blocks = 8192;
    cvt_f32_f16<<<dim3((unsigned)blocks), 256, 0, stream>>>(src, d, n);
    return d;
  };

  _Float16* Xh = cvtw(X, (long)16 * 512 * 768);
  _Float16* Wh0 = cvtw(P[h0_w], 768L * 768);
  _Float16* Wc0 = cvtw(P[c0_w], 768L * 768);
  _Float16* WihF = cvtw(P[wih_f], 3072L * 768);
  _Float16* WhhF = cvtw(P[whh_f], 3072L * 768);
  _Float16* WihR = cvtw(P[wih_r], 3072L * 768);
  _Float16* WhhR = cvtw(P[whh_r], 3072L * 768);
  _Float16* Wseqp = cvtw(P[seqp_w], 768L * 1536);
  _Float16* Wqkv = cvtw(P[qkv_w], 2304L * 768);
  _Float16* Wout = cvtw(P[out_w], 768L * 768);
  _Float16* Wm1 = cvtw(P[mlp1_w], 3072L * 768);
  _Float16* Wm2 = cvtw(P[mlp2_w], 768L * 3072);
  _Float16* Wf1 = cvtw(P[ffn1_w], 1536L * 768);
  _Float16* Wf2 = cvtw(P[ffn2_w], 768L * 1536);
  _Float16* Wzp = cvtw(P[zproj_w], 1536L * 768);
  _Float16* Wcp = cvtw(P[cp_w], 256L * 768);
  _Float16* Wzin = cvtw(P[zin_w], 256L * 1536);
  _Float16* Wzw1 = cvtw(P[zw1_w], 128L * 256);
  _Float16* Wzw2 = cvtw(P[zw2_w], 2L * 128);
  _Float16* Wcb = cvtw(P[cb_w], 768L * 256);
  _Float16* Wc1 = cvtw(P[c1_w], 256L * 768);
  _Float16* Wc2 = cvtw(P[c2_w], 8L * 256);

  float* bsumF = (float*)alloc(3072 * 4);
  float* bsumR = (float*)alloc(3072 * 4);
  float* h0f = (float*)alloc(16L * 768 * 4);
  float* c0f = (float*)alloc(16L * 768 * 4);
  _Float16* xpF = (_Float16*)alloc((size_t)8192 * 3072 * 2);
  _Float16* xpR = (_Float16*)alloc((size_t)8192 * 3072 * 2);
  _Float16* lstmO = (_Float16*)alloc((size_t)16 * 511 * 1536 * 2);
  _Float16* hT = (_Float16*)alloc(16L * 1536 * 2);
  float* xhat = (float*)alloc(16L * 768 * 4);
  float* Xf = (float*)alloc((size_t)8192 * 768 * 4);
  _Float16* hln = (_Float16*)alloc((size_t)8192 * 768 * 2);
  _Float16* qkvh = (_Float16*)alloc((size_t)8192 * 2304 * 2);
  _Float16* vT = (_Float16*)alloc((size_t)16 * 12 * 64 * 512 * 2);
  _Float16* oh = (_Float16*)alloc((size_t)8192 * 768 * 2);
  float* x1 = (float*)alloc((size_t)8192 * 768 * 4);
  _Float16* ln2h = (_Float16*)alloc((size_t)8192 * 768 * 2);
  _Float16* mid1 = (_Float16*)alloc((size_t)8192 * 3072 * 2);
  float* Af = (float*)alloc((size_t)8192 * 768 * 4);
  _Float16* Ah = (_Float16*)alloc((size_t)8192 * 768 * 2);
  _Float16* mid2 = (_Float16*)alloc((size_t)8192 * 1536 * 2);
  float* preo = (float*)alloc((size_t)8192 * 768 * 4);
  float* Xeen = (float*)alloc((size_t)8192 * 768 * 4);
  _Float16* Xeenh = (_Float16*)alloc((size_t)8192 * 768 * 2);
  _Float16* Zpre = (_Float16*)alloc(16L * 768 * 2);
  _Float16* Zh = (_Float16*)alloc(16L * 1536 * 2);
  _Float16* projh = (_Float16*)alloc(16L * 256 * 2);
  _Float16* h1h = (_Float16*)alloc(16L * 128 * 2);
  float* logits = (float*)alloc(16L * 2 * 4);
  float* om = (float*)alloc(16L * 2 * 4);
  float* V3 = (float*)alloc(48L * 256 * 4);
  float* Yc0 = (float*)alloc(16L * 256 * 4);
  _Float16* Yc0h = (_Float16*)alloc(16L * 256 * 2);
  float* Xconvp = (float*)alloc(16L * 768 * 4);
  float* Xconv = (float*)alloc(16L * 768 * 4);
  _Float16* Xf0h = (_Float16*)alloc(16L * 768 * 2);
  float* hcp = (float*)alloc(16L * 256 * 4);
  _Float16* hch = (_Float16*)alloc(16L * 256 * 2);

  const long BD = BIGDIV;

  vadd_kernel<<<12, 256, 0, stream>>>(P[bih_f], P[bhh_f], bsumF, 3072);
  vadd_kernel<<<12, 256, 0, stream>>>(P[bih_r], P[bhh_r], bsumR, 3072);

  gemm(stream, Xh, 768, 1, 512, 0, Wh0, 768, P[h0_b], nullptr, 0, h0f, nullptr,
       768, BD, 0, 0, 16, 768, 768, 0);
  gemm(stream, Xh, 768, 1, 512, 0, Wc0, 768, P[c0_b], nullptr, 0, c0f, nullptr,
       768, BD, 0, 0, 16, 768, 768, 0);

  gemm(stream, Xh, 768, BD, 0, 0, WihF, 768, bsumF, nullptr, 0, nullptr, xpF,
       3072, BD, 0, 0, 8192, 3072, 768, 0);
  gemm(stream, Xh, 768, BD, 0, 0, WihR, 768, bsumR, nullptr, 0, nullptr, xpR,
       3072, BD, 0, 0, 8192, 3072, 768, 0);

  lstm_scan_kernel<<<2, 1024, 270336, stream>>>(xpF, xpR, WhhF, WhhR, h0f, c0f,
                                                lstmO, hT);

  gemm(stream, lstmO, 1536, BD, 0, 0, Wseqp, 1536, P[seqp_b], nullptr, 0, Xf,
       nullptr, 768, 511, 512, 1, 16 * 511, 768, 1536, 0);
  gemm(stream, hT, 1536, BD, 0, 0, Wseqp, 1536, P[seqp_b], nullptr, 0, xhat,
       nullptr, 768, BD, 0, 0, 16, 768, 1536, 0);
  cls_gate_kernel<<<48, 256, 0, stream>>>(X, P[cls_gate], xhat, Xf);

  layernorm_kernel<<<1024, 256, 0, stream>>>(Xf, P[ln1_g], P[ln1_b], 8192, 768,
                                             nullptr, hln, 0);
  gemm(stream, hln, 768, BD, 0, 0, Wqkv, 768, P[qkv_b], nullptr, 0, nullptr, qkvh,
       2304, BD, 0, 0, 8192, 2304, 768, 0);
  vtrans_kernel<<<24576, 256, 0, stream>>>(qkvh, vT);
  attn_kernel<<<16 * 12 * 8, 128, 196608, stream>>>(qkvh, vT, mask, oh);
  gemm(stream, oh, 768, BD, 0, 0, Wout, 768, P[out_b], Xf, 768, x1, nullptr, 768,
       BD, 0, 0, 8192, 768, 768, 0);

  layernorm_kernel<<<1024, 256, 0, stream>>>(x1, P[ln2_g], P[ln2_b], 8192, 768,
                                             nullptr, ln2h, 0);
  gemm(stream, ln2h, 768, BD, 0, 0, Wm1, 768, P[mlp1_b], nullptr, 0, nullptr, mid1,
       3072, BD, 0, 0, 8192, 3072, 768, 2);
  gemm(stream, mid1, 3072, BD, 0, 0, Wm2, 3072, P[mlp2_b], x1, 768, Af, Ah, 768,
       BD, 0, 0, 8192, 768, 3072, 0);

  gemm(stream, Ah, 768, BD, 0, 0, Wf1, 768, P[ffn1_b], nullptr, 0, nullptr, mid2,
       1536, BD, 0, 0, 8192, 1536, 768, 2);
  gemm(stream, mid2, 1536, BD, 0, 0, Wf2, 1536, P[ffn2_b], Af, 768, preo, nullptr,
       768, BD, 0, 0, 8192, 768, 1536, 0);
  layernorm_kernel<<<1024, 256, 0, stream>>>(preo, P[lno_g], P[lno_b], 8192, 768,
                                             Xeen, Xeenh, 0);

  zprep_kernel<<<48, 256, 0, stream>>>(Xeen, P[alpha], Zpre);
  gemm(stream, Zpre, 768, BD, 0, 0, Wzp, 768, P[zproj_b], nullptr, 0, nullptr, Zh,
       1536, BD, 0, 0, 16, 1536, 768, 0);
  gemm(stream, Zh, 1536, BD, 0, 0, Wzin, 1536, P[zin_b], nullptr, 0, nullptr, projh,
       256, BD, 0, 0, 16, 256, 1536, 0);
  gemm(stream, projh, 256, BD, 0, 0, Wzw1, 256, P[zw1_b], nullptr, 0, nullptr, h1h,
       128, BD, 0, 0, 16, 128, 256, 1);
  gemm(stream, h1h, 128, BD, 0, 0, Wzw2, 128, P[zw2_b], nullptr, 0, logits, nullptr,
       2, BD, 0, 0, 16, 2, 128, 0);
  softmax2_kernel<<<1, 32, 0, stream>>>(logits, om);

  gemm(stream, Xeenh, 768, 3, 512, 0, Wcp, 768, P[cp_b], nullptr, 0, V3, nullptr,
       256, BD, 0, 0, 48, 256, 768, 1);
  yc0_kernel<<<16, 256, 0, stream>>>(V3, mask, om, P[wbank], P[bbank], Yc0, Yc0h);
  gemm(stream, Yc0h, 256, BD, 0, 0, Wcb, 256, P[cb_b], nullptr, 0, Xconvp, nullptr,
       768, BD, 0, 0, 16, 768, 256, 0);
  layernorm_kernel<<<2, 256, 0, stream>>>(Xconvp, P[lnf_g], P[lnf_b], 16, 768,
                                          Xconv, nullptr, 0);
  fuse0_kernel<<<48, 256, 0, stream>>>(Xeen, Xconv, P[fuse_gate], Xf0h);

  gemm(stream, Xf0h, 768, BD, 0, 0, Wc1, 768, P[c1_b], nullptr, 0, hcp, nullptr,
       256, BD, 0, 0, 16, 256, 768, 0);
  layernorm_kernel<<<2, 256, 0, stream>>>(hcp, P[cln_g], P[cln_b], 16, 256,
                                          nullptr, hch, 1);
  gemm(stream, hch, 256, BD, 0, 0, Wc2, 256, P[c2_b], nullptr, 0, (float*)d_out,
       nullptr, 8, BD, 0, 0, 16, 8, 256, 0);
  (void)Yc0; (void)hcp;
}
